// MSCrossAttnBlockDAUL1_5549097746876
// MI455X (gfx1250) — compile-verified, matrix operand order checked by reading, not run on hardware
//
#include <hip/hip_runtime.h>

// ---------------------------------------------------------------------------
// Types for CDNA5 WMMA (wave32): v16bf A/B fragments, v8f f32 accumulators
// ---------------------------------------------------------------------------
typedef __attribute__((ext_vector_type(16))) __bf16 v16bf;
typedef __attribute__((ext_vector_type(8)))  __bf16 v8bf;
typedef __attribute__((ext_vector_type(4)))  __bf16 v4bf;
typedef __attribute__((ext_vector_type(8)))  float  v8f;

#define ASTR 40  // LDS row stride (bf16 elems) = 80 B: 16B-aligned, bank-skewed

// ---------------------------------------------------------------------------
// GEMM: C[M x N] = A[M x 256] @ W[256 x N] + bias, N = NT*16, fp32 in/out,
// bf16 WMMA (v_wmma_f32_16x16x32_bf16) with fp32 accumulation.
// Block = 256 threads (8 waves); block tile = 128 rows x N cols.
// ---------------------------------------------------------------------------
template<int NT>
__global__ __launch_bounds__(256) void gemm_bias_k256(
    const float* __restrict__ A, const float* __restrict__ W,
    const float* __restrict__ bias, float* __restrict__ C, int M)
{
  (void)M;
  constexpr int N = NT * 16;
  __shared__ __bf16 As[128 * ASTR];
  __shared__ __bf16 Bs[N * ASTR];

  const int tid  = threadIdx.x;
  const int wid  = tid >> 5;
  const int lane = tid & 31;
  const int half = lane >> 4;
  const int ln   = lane & 15;
  const int k0   = half ? 8 : 0;   // ISA 16-bit A layout: lanes16-31 hold K=8..15/24..31
  const int m0   = blockIdx.x * 128;

  v8f acc[NT];
  #pragma unroll
  for (int nt = 0; nt < NT; ++nt)
    #pragma unroll
    for (int j = 0; j < 8; ++j) acc[nt][j] = 0.f;

  for (int kb = 0; kb < 256; kb += 32) {
    if (kb) __syncthreads();
    // ---- stage A: 128x32 fp32 -> bf16 LDS (row-major, padded stride) ----
    {
      const int kq = (tid & 7) * 4;
      const int r0 = tid >> 3;
      #pragma unroll
      for (int i = 0; i < 4; ++i) {
        const int r = r0 + 32 * i;
        const float4 v = *(const float4*)(A + (size_t)(m0 + r) * 256 + kb + kq);
        if (kb < 224)
          __builtin_prefetch(A + (size_t)(m0 + r) * 256 + kb + 32 + kq, 0, 1);
        v4bf pk;
        pk[0] = (__bf16)v.x; pk[1] = (__bf16)v.y;
        pk[2] = (__bf16)v.z; pk[3] = (__bf16)v.w;
        *(v4bf*)(&As[r * ASTR + kq]) = pk;
      }
    }
    // ---- stage B: W[32 x N] fp32 -> Bs[N][32] bf16 (transposed) ----
    {
      constexpr int NV = N / 4;
      for (int idx = tid; idx < 32 * NV; idx += 256) {
        const int k  = idx / NV;
        const int n4 = (idx % NV) * 4;
        const float4 v = *(const float4*)(W + (size_t)(kb + k) * N + n4);
        Bs[(n4 + 0) * ASTR + k] = (__bf16)v.x;
        Bs[(n4 + 1) * ASTR + k] = (__bf16)v.y;
        Bs[(n4 + 2) * ASTR + k] = (__bf16)v.z;
        Bs[(n4 + 3) * ASTR + k] = (__bf16)v.w;
      }
    }
    __syncthreads();

    // ---- A fragment: lane ln -> row, K split per ISA layout ----
    const __bf16* ap = &As[(wid * 16 + ln) * ASTR + k0];
    const v8bf alo = *(const v8bf*)(ap);        // K = k0 .. k0+7
    const v8bf ahi = *(const v8bf*)(ap + 16);   // K = k0+16 .. k0+23
    v16bf af;
    #pragma unroll
    for (int i = 0; i < 8; ++i) { af[i] = alo[i]; af[8 + i] = ahi[i]; }

    #pragma unroll
    for (int nt = 0; nt < NT; ++nt) {
      const __bf16* bp = &Bs[(nt * 16 + ln) * ASTR + k0];
      const v8bf blo = *(const v8bf*)(bp);
      const v8bf bhi = *(const v8bf*)(bp + 16);
      v16bf bfv;
      #pragma unroll
      for (int i = 0; i < 8; ++i) { bfv[i] = blo[i]; bfv[8 + i] = bhi[i]; }
      acc[nt] = __builtin_amdgcn_wmma_f32_16x16x32_bf16(
          false, af, false, bfv, (short)0, acc[nt], false, false);
    }
  }

  // ---- epilogue: bias + store (C/D layout: VGPR j -> M = j + 8*half) ----
  #pragma unroll
  for (int nt = 0; nt < NT; ++nt) {
    const int col = nt * 16 + ln;
    const float bv = bias[col];
    #pragma unroll
    for (int j = 0; j < 8; ++j) {
      const int row = m0 + wid * 16 + half * 8 + j;
      C[(size_t)row * N + col] = acc[nt][j] + bv;
    }
  }
}

// ---------------------------------------------------------------------------
// Row LayerNorm over 256 channels: one wave per row (wave32, 8 floats/lane).
// ---------------------------------------------------------------------------
__global__ __launch_bounds__(256) void ln256_k(
    const float* __restrict__ x, float* __restrict__ y,
    const float* __restrict__ g, const float* __restrict__ b, int rows)
{
  const int wid = threadIdx.x >> 5, lane = threadIdx.x & 31;
  const int row = blockIdx.x * 8 + wid;
  if (row >= rows) return;
  const float* xr = x + (size_t)row * 256;
  const int c = lane * 8;
  const float4 v0 = *(const float4*)(xr + c);
  const float4 v1 = *(const float4*)(xr + c + 4);
  float s  = v0.x + v0.y + v0.z + v0.w + v1.x + v1.y + v1.z + v1.w;
  float s2 = v0.x*v0.x + v0.y*v0.y + v0.z*v0.z + v0.w*v0.w
           + v1.x*v1.x + v1.y*v1.y + v1.z*v1.z + v1.w*v1.w;
  #pragma unroll
  for (int m = 16; m >= 1; m >>= 1) {
    s  += __shfl_xor(s,  m, 32);
    s2 += __shfl_xor(s2, m, 32);
  }
  const float mean = s * (1.f / 256.f);
  const float var  = s2 * (1.f / 256.f) - mean * mean;
  const float inv  = rsqrtf(var + 1e-6f);
  const float4 g0 = *(const float4*)(g + c), g1 = *(const float4*)(g + c + 4);
  const float4 b0 = *(const float4*)(b + c), b1 = *(const float4*)(b + c + 4);
  float4 o0, o1;
  o0.x = (v0.x - mean) * inv * g0.x + b0.x;
  o0.y = (v0.y - mean) * inv * g0.y + b0.y;
  o0.z = (v0.z - mean) * inv * g0.z + b0.z;
  o0.w = (v0.w - mean) * inv * g0.w + b0.w;
  o1.x = (v1.x - mean) * inv * g1.x + b1.x;
  o1.y = (v1.y - mean) * inv * g1.y + b1.y;
  o1.z = (v1.z - mean) * inv * g1.z + b1.z;
  o1.w = (v1.w - mean) * inv * g1.w + b1.w;
  *(float4*)(y + (size_t)row * 256 + c)     = o0;
  *(float4*)(y + (size_t)row * 256 + c + 4) = o1;
}

// ---------------------------------------------------------------------------
// Softmax over NP=4 (in place).
// ---------------------------------------------------------------------------
__global__ void softmax4_k(float* __restrict__ x, int groups)
{
  const int i = blockIdx.x * blockDim.x + threadIdx.x;
  if (i >= groups) return;
  float4 v = *(float4*)(x + (size_t)i * 4);
  const float m  = fmaxf(fmaxf(v.x, v.y), fmaxf(v.z, v.w));
  const float e0 = __expf(v.x - m), e1 = __expf(v.y - m);
  const float e2 = __expf(v.z - m), e3 = __expf(v.w - m);
  const float r  = 1.f / (e0 + e1 + e2 + e3);
  const float4 o = { e0 * r, e1 * r, e2 * r, e3 * r };
  *(float4*)(x + (size_t)i * 4) = o;
}

// ---------------------------------------------------------------------------
// MSDeformAttn bilinear sampling + attention-weight combine.
// Thread = (n,q,h,d); 16 lanes (d) per head -> coalesced 64B value gathers.
// Query grid is always 32x32 (Lq=1024); `e` is the feature-level size.
// off: (N*1024, 128) [h*8+p*2+{x,y}]; aw: (N*1024, 64) [h*4+p] (softmaxed);
// value: (N, e*e, 256); out: (N*1024, 256).
// ---------------------------------------------------------------------------
__global__ __launch_bounds__(256) void msda_sample_k(
    const float* __restrict__ off, const float* __restrict__ aw,
    const float* __restrict__ value, float* __restrict__ out, int e)
{
  const int t = blockIdx.x * 256 + threadIdx.x;
  const int d = t & 15;
  const int h = (t >> 4) & 15;
  const int q = (t >> 8) & 1023;
  const int n = t >> 18;
  const int row = (n << 10) | q;
  const float rx = ((q & 31) + 0.5f) * (1.f / 32.f);
  const float ry = ((q >> 5) + 0.5f) * (1.f / 32.f);
  const float fe = (float)e;
  const float inv_e = 1.f / fe;
  const float* vb = value + (size_t)n * e * e * 256 + h * 16 + d;
  const float* op = off + (size_t)row * 128 + h * 8;
  const float* ap = aw  + (size_t)row * 64  + h * 4;
  float acc = 0.f;
  #pragma unroll
  for (int p = 0; p < 4; ++p) {
    const float xf = (rx + op[2 * p]     * inv_e) * fe - 0.5f;
    const float yf = (ry + op[2 * p + 1] * inv_e) * fe - 0.5f;
    const float x0 = floorf(xf), y0 = floorf(yf);
    const float fx = xf - x0,    fy = yf - y0;
    const int ix = (int)x0, iy = (int)y0;
    const float a = ap[p];
    const float w00 = (1.f - fx) * (1.f - fy), w10 = fx * (1.f - fy);
    const float w01 = (1.f - fx) * fy,         w11 = fx * fy;
    const bool vx0 = (ix >= 0) & (ix < e),     vx1 = (ix + 1 >= 0) & (ix + 1 < e);
    const bool vy0 = (iy >= 0) & (iy < e),     vy1 = (iy + 1 >= 0) & (iy + 1 < e);
    if (vx0 & vy0) acc += a * w00 * vb[(size_t)(iy       * e + ix    ) * 256];
    if (vx1 & vy0) acc += a * w10 * vb[(size_t)(iy       * e + ix + 1) * 256];
    if (vx0 & vy1) acc += a * w01 * vb[(size_t)((iy + 1) * e + ix    ) * 256];
    if (vx1 & vy1) acc += a * w11 * vb[(size_t)((iy + 1) * e + ix + 1) * 256];
  }
  out[(size_t)row * 256 + h * 16 + d] = acc;
}

// ---------------------------------------------------------------------------
// Elementwise combines.
// ---------------------------------------------------------------------------
__global__ void scale_add2_k(const float* __restrict__ a, const float* __restrict__ b,
                             const float* __restrict__ gamma, float* __restrict__ o, int n)
{
  const int i = blockIdx.x * blockDim.x + threadIdx.x;
  if (i < n) o[i] = gamma[i & 255] * (a[i] + b[i]);
}

__global__ void add4_k(const float* __restrict__ s, const float* __restrict__ a,
                       const float* __restrict__ b, const float* __restrict__ c,
                       float* __restrict__ o, int n)
{
  const int i = blockIdx.x * blockDim.x + threadIdx.x;
  if (i < n) o[i] = s[i] + a[i] + b[i] + c[i];
}

// ---------------------------------------------------------------------------
// Host-side orchestration
// ---------------------------------------------------------------------------
struct MsdaP {
  const float *off_w, *off_b, *aw_w, *aw_b, *vp_w, *vp_b, *op_w, *op_b;
};
struct CsaP {
  MsdaP cross, self;
  const float *qn_w, *qn_b, *qn_g, *qn_beta;
  const float *fn_w, *fn_b, *fn_g, *fn_beta;
  const float *norm_g, *norm_b, *gamma1, *gamma2;
};

static void run_csa(hipStream_t s, const CsaP& P,
                    const float* query, const float* feat, int e,
                    float* fn, float* val, float* qn, float* off, float* aw,
                    float* samp, float* attn, float* attn1, float* attn2,
                    float* csa_out)
{
  const int Mq = 8192;            // 8 batches x 1024 query tokens
  const int Mf = 8 * e * e;       // 8 batches x e*e feature tokens
  const dim3 blk(256);

  // qn = LN(query @ qn_w + qn_b)
  gemm_bias_k256<16><<<Mq / 128, blk, 0, s>>>(query, P.qn_w, P.qn_b, qn, Mq);
  ln256_k<<<Mq / 8, blk, 0, s>>>(qn, qn, P.qn_g, P.qn_beta, Mq);
  // fn = LN(feat @ fn_w + fn_b)
  gemm_bias_k256<16><<<Mf / 128, blk, 0, s>>>(feat, P.fn_w, P.fn_b, fn, Mf);
  ln256_k<<<Mf / 8, blk, 0, s>>>(fn, fn, P.fn_g, P.fn_beta, Mf);

  // ---- cross deformable attention ----
  gemm_bias_k256<16><<<Mf / 128, blk, 0, s>>>(fn, P.cross.vp_w, P.cross.vp_b, val, Mf);
  gemm_bias_k256<8><<<Mq / 128, blk, 0, s>>>(qn, P.cross.off_w, P.cross.off_b, off, Mq);
  gemm_bias_k256<4><<<Mq / 128, blk, 0, s>>>(qn, P.cross.aw_w, P.cross.aw_b, aw, Mq);
  softmax4_k<<<(Mq * 16) / 256, blk, 0, s>>>(aw, Mq * 16);
  msda_sample_k<<<8192, blk, 0, s>>>(off, aw, val, samp, e);
  gemm_bias_k256<16><<<Mq / 128, blk, 0, s>>>(samp, P.cross.op_w, P.cross.op_b, attn, Mq);

  // attn1 = LN(attn)
  ln256_k<<<Mq / 8, blk, 0, s>>>(attn, attn1, P.norm_g, P.norm_b, Mq);

  // ---- self deformable attention (query = feat = attn1, e = 32) ----
  gemm_bias_k256<16><<<Mq / 128, blk, 0, s>>>(attn1, P.self.vp_w, P.self.vp_b, val, Mq);
  gemm_bias_k256<8><<<Mq / 128, blk, 0, s>>>(attn1, P.self.off_w, P.self.off_b, off, Mq);
  gemm_bias_k256<4><<<Mq / 128, blk, 0, s>>>(attn1, P.self.aw_w, P.self.aw_b, aw, Mq);
  softmax4_k<<<(Mq * 16) / 256, blk, 0, s>>>(aw, Mq * 16);
  msda_sample_k<<<8192, blk, 0, s>>>(off, aw, val, samp, 32);
  gemm_bias_k256<16><<<Mq / 128, blk, 0, s>>>(samp, P.self.op_w, P.self.op_b, attn2, Mq);

  // csa_out = gamma1 * (attn + attn2)
  scale_add2_k<<<(Mq * 256) / 256, blk, 0, s>>>(attn, attn2, P.gamma1, csa_out, Mq * 256);
}

extern "C" void kernel_launch(void* const* d_in, const int* in_sizes, int n_in,
                              void* d_out, int out_size, void* d_ws, size_t ws_size,
                              hipStream_t stream)
{
  (void)in_sizes; (void)n_in; (void)out_size; (void)ws_size;

  const float* src0     = (const float*)d_in[0];
  const float* src_res0 = (const float*)d_in[1];
  const float* src1     = (const float*)d_in[2];
  const float* src_res1 = (const float*)d_in[3];
  const float* src2     = (const float*)d_in[4];
  const float* src_res2 = (const float*)d_in[5];

  // params in setup_inputs() dict-insertion order:
  // ca, ca1, ca2, car, car1, car2; each: cross{off_w,off_b,aw_w,aw_b,vp_w,vp_b,
  // op_w,op_b}, self{...}, qn_w,qn_b,qn_g,qn_beta, fn_w,fn_b,fn_g,fn_beta,
  // norm_g,norm_b, gamma1,gamma2
  CsaP P[6];
  int idx = 6;
  auto F = [&]() -> const float* { return (const float*)d_in[idx++]; };
  for (int c = 0; c < 6; ++c) {
    MsdaP* ms[2] = { &P[c].cross, &P[c].self };
    for (int m = 0; m < 2; ++m) {
      ms[m]->off_w = F(); ms[m]->off_b = F();
      ms[m]->aw_w  = F(); ms[m]->aw_b  = F();
      ms[m]->vp_w  = F(); ms[m]->vp_b  = F();
      ms[m]->op_w  = F(); ms[m]->op_b  = F();
    }
    P[c].qn_w = F(); P[c].qn_b = F(); P[c].qn_g = F(); P[c].qn_beta = F();
    P[c].fn_w = F(); P[c].fn_b = F(); P[c].fn_g = F(); P[c].fn_beta = F();
    P[c].norm_g = F(); P[c].norm_b = F();
    P[c].gamma1 = F(); P[c].gamma2 = F();
  }

  // ---- workspace carve (deterministic, ~370 MB) ----
  char* w = (char*)d_ws;
  auto carve = [&](size_t bytes) -> float* {
    float* p = (float*)w;
    w += (bytes + 255) & ~(size_t)255;
    return p;
  };
  float* fn    = carve((size_t)8 * 16384 * 256 * 4);
  float* val   = carve((size_t)8 * 16384 * 256 * 4);
  float* qn    = carve((size_t)8192 * 256 * 4);
  float* off   = carve((size_t)8192 * 128 * 4);
  float* aw    = carve((size_t)8192 * 64 * 4);
  float* samp  = carve((size_t)8192 * 256 * 4);
  float* attn  = carve((size_t)8192 * 256 * 4);
  float* attn1 = carve((size_t)8192 * 256 * 4);
  float* attn2 = carve((size_t)8192 * 256 * 4);
  float* cout[6];
  for (int i = 0; i < 6; ++i) cout[i] = carve((size_t)8192 * 256 * 4);

  // P index: 0=ca, 1=ca1, 2=ca2, 3=car, 4=car1, 5=car2
  run_csa(stream, P[0], src2,     src_res0, 128, fn, val, qn, off, aw, samp, attn, attn1, attn2, cout[0]);
  run_csa(stream, P[3], src_res2, src0,     128, fn, val, qn, off, aw, samp, attn, attn1, attn2, cout[3]);
  run_csa(stream, P[1], cout[0],  src_res1,  64, fn, val, qn, off, aw, samp, attn, attn1, attn2, cout[1]);
  run_csa(stream, P[4], cout[3],  src1,      64, fn, val, qn, off, aw, samp, attn, attn1, attn2, cout[4]);
  run_csa(stream, P[2], cout[1],  src_res2,  32, fn, val, qn, off, aw, samp, attn, attn1, attn2, cout[2]);
  run_csa(stream, P[5], cout[4],  src2,      32, fn, val, qn, off, aw, samp, attn, attn1, attn2, cout[5]);

  float* outp = (float*)d_out;
  const int NQ = 8192 * 256;
  add4_k<<<NQ / 256, 256, 0, stream>>>(src2,     cout[0], cout[1], cout[2], outp,      NQ);
  add4_k<<<NQ / 256, 256, 0, stream>>>(src_res2, cout[3], cout[4], cout[5], outp + NQ, NQ);
}
